// Net3DVAE_31885837205715
// MI455X (gfx1250) — matrix operand order repaired and checked by reading, not compiled, not run on hardware
//
#include <hip/hip_runtime.h>
#include <hip/hip_bf16.h>

#define HDIM 128
#define NE   262144
#define NN   32768
#define NB   256
#define NL   4
#define NFE  4

typedef __attribute__((ext_vector_type(16))) _Float16 v16h;
typedef __attribute__((ext_vector_type(8)))  _Float16 v8h;
typedef __attribute__((ext_vector_type(8)))  float    v8f;

__device__ __forceinline__ float silu_f(float x)    { return x / (1.0f + __expf(-x)); }
__device__ __forceinline__ float sigmoid_f(float x) { return 1.0f / (1.0f + __expf(-x)); }

// A-fragment (16x32 f16, M=row per lane&15) per CDNA5 ISA layout:
// lanes 0-15: elems 0-7 = K k+0..7,  elems 8-15 = K k+16..23
// lanes 16-31: elems 0-7 = K k+8..15, elems 8-15 = K k+24..31
__device__ __forceinline__ v16h load_a_frag(const _Float16* rowPtr, int k, int lane) {
  const int hi8 = (lane & 16) ? 8 : 0;
  v8h lo = *(const v8h*)(rowPtr + k + hi8);
  v8h hi = *(const v8h*)(rowPtr + k + 16 + hi8);
  return __builtin_shufflevector(lo, hi, 0,1,2,3,4,5,6,7,8,9,10,11,12,13,14,15);
}

// B-fragment (32x16 f16): lane holds column lane&15; lanes 0-15: elems 0-7 = K k+0..7,
// elems 8-15 = K k+16..23; lanes 16-31: K k+8..15, k+24..31.  Weights stored transposed: wt[col][K].
__device__ __forceinline__ v16h load_b_frag(const _Float16* wt, int K, int col, int k, int lane) {
  const int hi8 = (lane & 16) ? 8 : 0;
  const _Float16* p = wt + (size_t)col * K + k;
  v8h lo = *(const v8h*)(p + hi8);
  v8h hi = *(const v8h*)(p + 16 + hi8);
  return __builtin_shufflevector(lo, hi, 0,1,2,3,4,5,6,7,8,9,10,11,12,13,14,15);
}

__device__ __forceinline__ void atomic_max_f32(float* addr, float val) {
  unsigned int* ua = (unsigned int*)addr;
  unsigned int old = *ua;
  while (__uint_as_float(old) < val) {
    unsigned int assumed = old;
    old = atomicCAS(ua, assumed, __float_as_uint(val));
    if (old == assumed) break;
  }
}

// ---------------- prep kernels ----------------

__global__ __launch_bounds__(256) void k_transpose_f16(const float* __restrict__ src,
    _Float16* __restrict__ dst, int rows, int cols, long total) {
  long t = (long)blockIdx.x * 256 + threadIdx.x;
  if (t >= total) return;
  long per = (long)rows * cols;
  int b = (int)(t / per);
  long rem = t - (long)b * per;
  int r = (int)(rem / cols);
  int c = (int)(rem - (long)r * cols);
  dst[(size_t)b * per + (size_t)c * rows + r] = (_Float16)src[t];
}

__global__ __launch_bounds__(256) void k_fill_f32(float* __restrict__ p, float v, long n) {
  long t = (long)blockIdx.x * 256 + threadIdx.x;
  if (t < n) p[t] = v;
}

__global__ __launch_bounds__(256) void k_edge_encode(const float* __restrict__ d,
    const float* __restrict__ We, const float* __restrict__ be,
    float* __restrict__ e_f, _Float16* __restrict__ e_h) {
  long t = (long)blockIdx.x * 256 + threadIdx.x;      // E*H threads
  int e = (int)(t >> 7);
  int h = (int)(t & (HDIM - 1));
  float dv = d[e];
  float de[2 * NFE + 1];
#pragma unroll
  for (int j = 0; j < NFE; ++j) {
    float x = dv / (float)(1 << j);
    de[j]       = __sinf(x);
    de[NFE + j] = __cosf(x);
  }
  de[2 * NFE] = dv;
  float acc = be[h];
#pragma unroll
  for (int j = 0; j < 2 * NFE + 1; ++j) acc += de[j] * We[j * HDIM + h];
  float v = silu_f(silu_f(acc));
  e_f[t] = v;
  e_h[t] = (_Float16)v;
}

__global__ __launch_bounds__(256) void k_feat_init(const float* __restrict__ node_emb,
    float* __restrict__ feat_f, _Float16* __restrict__ feat_h) {
  long t = (long)blockIdx.x * 256 + threadIdx.x;      // N*H threads
  int h = (int)(t & (HDIM - 1));
  float v = node_emb[h];
  feat_f[t] = v;
  feat_h[t] = (_Float16)v;
}

// ---------------- edge message kernel (WMMA) ----------------
// 1 block = 16 edges, 256 threads = 8 waves; wave w owns output columns [16w,16w+16).
// mi = [feat[src] | feat[dst] | e] is staged once into LDS (coalesced), then all
// waves read A-fragments via ds_load — 8x less global gather traffic.
#define LDK (3 * HDIM + 8)     // 392 halves: rows 16B-aligned, stride 196 dwords == 4 mod 64
__global__ __launch_bounds__(256) void k_edge_msg(int l,
    const _Float16* __restrict__ feat_h,
    float* __restrict__ e_f, _Float16* e_h,
    const int* __restrict__ src, const int* __restrict__ dst,
    const _Float16* __restrict__ W1t, const float* __restrict__ b1,
    const _Float16* __restrict__ W2t, const float* __restrict__ b2,
    const float* __restrict__ softW, const float* __restrict__ softb,
    float* __restrict__ msum) {
  const int tid      = threadIdx.x;
  const int lane     = tid & 31;
  const int wave     = tid >> 5;
  const int colBase  = wave * 16;
  const int tileBase = blockIdx.x * 16;
  const int mcol     = lane & 15;
  const int col      = colBase + mcol;
  const int rbase    = (lane & 16) ? 8 : 0;

  __shared__ __align__(16) _Float16 mi_lds[16][LDK];
  __shared__ __align__(16) _Float16 hid_lds[16][HDIM + 8];
  __shared__ float m_lds[16][HDIM + 4];
  __shared__ float w_lds[16];

  // ---- stage mi tile: thread t -> edge t>>4, 8-half chunk (t&15)*8 per segment
  {
    const int el = tid >> 4;
    const int hb = (tid & 15) * 8;
    const int grow = tileBase + el;
    const int sN = src[grow];
    const int dN = dst[grow];
    *(v8h*)&mi_lds[el][hb]            = *(const v8h*)(feat_h + (size_t)sN * HDIM + hb);
    *(v8h*)&mi_lds[el][HDIM + hb]     = *(const v8h*)(feat_h + (size_t)dN * HDIM + hb);
    *(v8h*)&mi_lds[el][2 * HDIM + hb] = *(const v8h*)(e_h    + (size_t)grow * HDIM + hb);
  }
  __syncthreads();

  const _Float16* W1 = W1t + (size_t)l * HDIM * (3 * HDIM);
  const _Float16* W2 = W2t + (size_t)l * HDIM * HDIM;

  // GEMM1: hid[16 x col-tile] = mi @ msg_W1
  v8f acc = {};
#pragma unroll
  for (int kk = 0; kk < 3 * HDIM; kk += 32) {
    v16h a = load_a_frag(&mi_lds[mcol][0], kk, lane);
    v16h b = load_b_frag(W1, 3 * HDIM, col, kk, lane);
    acc = __builtin_amdgcn_wmma_f32_16x16x32_f16(false, a, false, b, (short)0, acc, false, false);
  }
  {
    const float bias = b1[l * HDIM + col];
#pragma unroll
    for (int i = 0; i < 8; ++i)
      hid_lds[rbase + i][col] = (_Float16)silu_f(acc[i] + bias);
  }
  __syncthreads();

  // GEMM2: m = silu(hid @ msg_W2 + b2)
  v8f acc2 = {};
#pragma unroll
  for (int kk = 0; kk < HDIM; kk += 32) {
    v16h a = load_a_frag(&hid_lds[mcol][0], kk, lane);
    v16h b = load_b_frag(W2, HDIM, col, kk, lane);
    acc2 = __builtin_amdgcn_wmma_f32_16x16x32_f16(false, a, false, b, (short)0, acc2, false, false);
  }
  {
    const float bias = b2[l * HDIM + col];
#pragma unroll
    for (int i = 0; i < 8; ++i)
      m_lds[rbase + i][col] = silu_f(acc2[i] + bias);
  }
  __syncthreads();

  // soft edge gate: w = sigmoid(m . soft_W + soft_b)
  if (tid < 16) {
    float g = softb[l];
    const float* sw = softW + l * HDIM;
    for (int h = 0; h < HDIM; ++h) g += m_lds[tid][h] * sw[h];
    w_lds[tid] = sigmoid_f(g);
  }
  __syncthreads();

  // e += m ; scatter m*w into msum[dst]
  {
    const int el = tid >> 4;
    const int hb = (tid & 15) * 8;
    const int grow = tileBase + el;
    const float w = w_lds[el];
    const int dn = dst[grow];
    float*     msumRow = msum + (size_t)dn * HDIM;
    float*     eRow    = e_f  + (size_t)grow * HDIM;
    _Float16*  ehRow   = e_h  + (size_t)grow * HDIM;
#pragma unroll
    for (int i = 0; i < 8; ++i) {
      const int h = hb + i;
      const float mv = m_lds[el][h];
      const float ev = eRow[h] + mv;
      eRow[h] = ev;
      ehRow[h] = (_Float16)ev;
      atomicAdd(msumRow + h, mv * w);
    }
  }
}

// ---------------- node update kernel (WMMA) ----------------
__global__ __launch_bounds__(256) void k_node_upd(int l,
    float* feat_f, _Float16* feat_h,
    const float* __restrict__ msum,
    const _Float16* __restrict__ W1t, const float* __restrict__ b1,
    const _Float16* __restrict__ W2t, const float* __restrict__ b2) {
  const int tid      = threadIdx.x;
  const int lane     = tid & 31;
  const int wave     = tid >> 5;
  const int colBase  = wave * 16;
  const int tileBase = blockIdx.x * 16;
  const int mcol     = lane & 15;
  const int col      = colBase + mcol;
  const int rbase    = (lane & 16) ? 8 : 0;

  __shared__ __align__(16) _Float16 uin_lds[16][HDIM + 8];
  __shared__ __align__(16) _Float16 hid_lds[16][HDIM + 8];

  for (int t = tid; t < 16 * HDIM; t += 256) {
    int r = t >> 7, h = t & (HDIM - 1);
    size_t g = (size_t)(tileBase + r) * HDIM + h;
    uin_lds[r][h] = (_Float16)(msum[g] + feat_f[g]);
  }
  __syncthreads();

  const _Float16* W1 = W1t + (size_t)l * HDIM * HDIM;
  const _Float16* W2 = W2t + (size_t)l * HDIM * HDIM;

  v8f acc = {};
#pragma unroll
  for (int kk = 0; kk < HDIM; kk += 32) {
    v16h a = load_a_frag(&uin_lds[mcol][0], kk, lane);
    v16h b = load_b_frag(W1, HDIM, col, kk, lane);
    acc = __builtin_amdgcn_wmma_f32_16x16x32_f16(false, a, false, b, (short)0, acc, false, false);
  }
  {
    const float bias = b1[l * HDIM + col];
#pragma unroll
    for (int i = 0; i < 8; ++i)
      hid_lds[rbase + i][col] = (_Float16)silu_f(acc[i] + bias);
  }
  __syncthreads();

  v8f acc2 = {};
#pragma unroll
  for (int kk = 0; kk < HDIM; kk += 32) {
    v16h a = load_a_frag(&hid_lds[mcol][0], kk, lane);
    v16h b = load_b_frag(W2, HDIM, col, kk, lane);
    acc2 = __builtin_amdgcn_wmma_f32_16x16x32_f16(false, a, false, b, (short)0, acc2, false, false);
  }
  {
    const float bias = b2[l * HDIM + col];
#pragma unroll
    for (int i = 0; i < 8; ++i) {
      size_t g = (size_t)(tileBase + rbase + i) * HDIM + col;
      float v = acc2[i] + bias + feat_f[g];
      feat_f[g] = v;
      feat_h[g] = (_Float16)v;
    }
  }
}

// ---------------- output MLP kernel (WMMA) ----------------
__global__ __launch_bounds__(256) void k_out_net(
    const _Float16* __restrict__ feat_h,
    const _Float16* __restrict__ W1t, const float* __restrict__ b1,
    const _Float16* __restrict__ W2t, const float* __restrict__ b2,
    float* __restrict__ nodeout) {
  const int tid      = threadIdx.x;
  const int lane     = tid & 31;
  const int wave     = tid >> 5;
  const int colBase  = wave * 16;
  const int tileBase = blockIdx.x * 16;
  const int mcol     = lane & 15;
  const int col      = colBase + mcol;
  const int rbase    = (lane & 16) ? 8 : 0;

  __shared__ __align__(16) _Float16 fin_lds[16][HDIM + 8];
  __shared__ __align__(16) _Float16 hid_lds[16][HDIM + 8];

  // stage feat tile once (coalesced), waves read A-fragments from LDS
  {
    const int r  = tid >> 4;
    const int hb = (tid & 15) * 8;
    *(v8h*)&fin_lds[r][hb] = *(const v8h*)(feat_h + (size_t)(tileBase + r) * HDIM + hb);
  }
  __syncthreads();

  v8f acc = {};
#pragma unroll
  for (int kk = 0; kk < HDIM; kk += 32) {
    v16h a = load_a_frag(&fin_lds[mcol][0], kk, lane);
    v16h b = load_b_frag(W1t, HDIM, col, kk, lane);
    acc = __builtin_amdgcn_wmma_f32_16x16x32_f16(false, a, false, b, (short)0, acc, false, false);
  }
  {
    const float bias = b1[col];
#pragma unroll
    for (int i = 0; i < 8; ++i)
      hid_lds[rbase + i][col] = (_Float16)silu_f(acc[i] + bias);
  }
  __syncthreads();

  v8f acc2 = {};
#pragma unroll
  for (int kk = 0; kk < HDIM; kk += 32) {
    v16h a = load_a_frag(&hid_lds[mcol][0], kk, lane);
    v16h b = load_b_frag(W2t, HDIM, col, kk, lane);
    acc2 = __builtin_amdgcn_wmma_f32_16x16x32_f16(false, a, false, b, (short)0, acc2, false, false);
  }
  {
    const float bias = b2[col];
#pragma unroll
    for (int i = 0; i < 8; ++i)
      nodeout[(size_t)(tileBase + rbase + i) * HDIM + col] = acc2[i] + bias;
  }
}

// ---------------- readout ----------------
__global__ __launch_bounds__(256) void k_readout_acc(const float* __restrict__ nodeout,
    const int* __restrict__ seg, float* __restrict__ s, float* __restrict__ mx,
    float* __restrict__ cnt) {
  long t = (long)blockIdx.x * 256 + threadIdx.x;       // N*H threads
  int n = (int)(t >> 7);
  int h = (int)(t & (HDIM - 1));
  int b = seg[n];
  float v = nodeout[t];
  atomicAdd(&s[(size_t)b * HDIM + h], v);
  atomic_max_f32(&mx[(size_t)b * HDIM + h], v);
  if (h == 0) atomicAdd(&cnt[b], 1.0f);
}

__global__ __launch_bounds__(256) void k_readout_write(const float* __restrict__ s,
    const float* __restrict__ mx, const float* __restrict__ cnt, float* __restrict__ out) {
  long t = (long)blockIdx.x * 256 + threadIdx.x;       // B*H threads
  int b = (int)(t >> 7);
  int h = (int)(t & (HDIM - 1));
  float sv = s[t];
  float c = fmaxf(cnt[b], 1.0f);
  float* o = out + (size_t)b * 3 * HDIM;
  o[h]            = sv;
  o[HDIM + h]     = sv / c;
  o[2 * HDIM + h] = mx[t];
}

// ---------------- launch ----------------
extern "C" void kernel_launch(void* const* d_in, const int* in_sizes, int n_in,
                              void* d_out, int out_size, void* d_ws, size_t ws_size,
                              hipStream_t stream) {
  (void)in_sizes; (void)n_in; (void)out_size; (void)ws_size;
  const float* d        = (const float*)d_in[0];
  const int*   src      = (const int*)d_in[1];
  const int*   dst      = (const int*)d_in[2];
  const int*   seg      = (const int*)d_in[3];
  const float* node_emb = (const float*)d_in[4];
  const float* We       = (const float*)d_in[5];
  const float* be       = (const float*)d_in[6];
  const float* msg_W1   = (const float*)d_in[7];
  const float* msg_b1   = (const float*)d_in[8];
  const float* msg_W2   = (const float*)d_in[9];
  const float* msg_b2   = (const float*)d_in[10];
  const float* soft_W   = (const float*)d_in[11];
  const float* soft_b   = (const float*)d_in[12];
  const float* upd_W1   = (const float*)d_in[13];
  const float* upd_b1   = (const float*)d_in[14];
  const float* upd_W2   = (const float*)d_in[15];
  const float* upd_b2   = (const float*)d_in[16];
  const float* out_W1   = (const float*)d_in[17];
  const float* out_b1   = (const float*)d_in[18];
  const float* out_W2   = (const float*)d_in[19];
  const float* out_b2   = (const float*)d_in[20];
  float* out = (float*)d_out;

  char* ws = (char*)d_ws;
  size_t off = 0;
  auto take = [&](size_t bytes) -> void* {
    void* p = ws + off;
    off += (bytes + 255) & ~(size_t)255;
    return p;
  };
  float*    e_f    = (float*)   take((size_t)NE * HDIM * 4);
  _Float16* e_h    = (_Float16*)take((size_t)NE * HDIM * 2);
  float*    feat_f = (float*)   take((size_t)NN * HDIM * 4);
  _Float16* feat_h = (_Float16*)take((size_t)NN * HDIM * 2);
  float*    msum   = (float*)   take((size_t)NN * HDIM * 4);   // reused as nodeout after layers
  _Float16* msgW1t = (_Float16*)take((size_t)NL * 3 * HDIM * HDIM * 2);
  _Float16* msgW2t = (_Float16*)take((size_t)NL * HDIM * HDIM * 2);
  _Float16* updW1t = (_Float16*)take((size_t)NL * HDIM * HDIM * 2);
  _Float16* updW2t = (_Float16*)take((size_t)NL * HDIM * HDIM * 2);
  _Float16* outW1t = (_Float16*)take((size_t)HDIM * HDIM * 2);
  _Float16* outW2t = (_Float16*)take((size_t)HDIM * HDIM * 2);
  float*    s_acc  = (float*)   take((size_t)NB * HDIM * 4);
  float*    mx_acc = (float*)   take((size_t)NB * HDIM * 4);
  float*    cnt    = (float*)   take((size_t)NB * 4);

  auto grid1 = [](long n) { return dim3((unsigned)((n + 255) / 256)); };

  // weight transposes (fp32 -> f16 [N][K])
  k_transpose_f16<<<grid1((long)NL * 3 * HDIM * HDIM), 256, 0, stream>>>(msg_W1, msgW1t, 3 * HDIM, HDIM, (long)NL * 3 * HDIM * HDIM);
  k_transpose_f16<<<grid1((long)NL * HDIM * HDIM), 256, 0, stream>>>(msg_W2, msgW2t, HDIM, HDIM, (long)NL * HDIM * HDIM);
  k_transpose_f16<<<grid1((long)NL * HDIM * HDIM), 256, 0, stream>>>(upd_W1, updW1t, HDIM, HDIM, (long)NL * HDIM * HDIM);
  k_transpose_f16<<<grid1((long)NL * HDIM * HDIM), 256, 0, stream>>>(upd_W2, updW2t, HDIM, HDIM, (long)NL * HDIM * HDIM);
  k_transpose_f16<<<grid1((long)HDIM * HDIM), 256, 0, stream>>>(out_W1, outW1t, HDIM, HDIM, (long)HDIM * HDIM);
  k_transpose_f16<<<grid1((long)HDIM * HDIM), 256, 0, stream>>>(out_W2, outW2t, HDIM, HDIM, (long)HDIM * HDIM);

  // edge features + node feature init
  k_edge_encode<<<grid1((long)NE * HDIM), 256, 0, stream>>>(d, We, be, e_f, e_h);
  k_feat_init<<<grid1((long)NN * HDIM), 256, 0, stream>>>(node_emb, feat_f, feat_h);

  // readout accumulators
  k_fill_f32<<<grid1((long)NB * HDIM), 256, 0, stream>>>(s_acc, 0.0f, (long)NB * HDIM);
  k_fill_f32<<<grid1((long)NB * HDIM), 256, 0, stream>>>(mx_acc, -INFINITY, (long)NB * HDIM);
  k_fill_f32<<<grid1((long)NB), 256, 0, stream>>>(cnt, 0.0f, (long)NB);

  for (int l = 0; l < NL; ++l) {
    k_fill_f32<<<grid1((long)NN * HDIM), 256, 0, stream>>>(msum, 0.0f, (long)NN * HDIM);
    k_edge_msg<<<dim3(NE / 16), 256, 0, stream>>>(l, feat_h, e_f, e_h, src, dst,
        msgW1t, msg_b1, msgW2t, msg_b2, soft_W, soft_b, msum);
    k_node_upd<<<dim3(NN / 16), 256, 0, stream>>>(l, feat_f, feat_h, msum,
        updW1t, upd_b1, updW2t, upd_b2);
  }

  k_out_net<<<dim3(NN / 16), 256, 0, stream>>>(feat_h, outW1t, out_b1, outW2t, out_b2, msum);
  k_readout_acc<<<grid1((long)NN * HDIM), 256, 0, stream>>>(msum, seg, s_acc, mx_acc, cnt);
  k_readout_write<<<grid1((long)NB * HDIM), 256, 0, stream>>>(s_acc, mx_acc, cnt, out);
}
